// CausalAttention_47751446397549
// MI455X (gfx1250) — compile-verified, matrix operand order checked by reading, not run on hardware
//
#include <hip/hip_runtime.h>

// ---------------------------------------------------------------------------
// Causal attention, B=4, N=2048, D_IN=D_OUT=1024, fp32 in/out.
//   1) fp32 -> bf16 conversion pass (x, Wq, Wk, Wv)
//   2) QKV projections via v_wmma_f32_16x16x32_bf16 (x tile staged by TDM,
//      V stored transposed)
//   3) flash attention: TDM stages the Q tile into LDS, S = Q K^T and
//      O += P V via WMMA, ds_add_f32 cross-wave S reduction, online softmax,
//      global_prefetch for the next key tile, fp32 accumulation.
// ---------------------------------------------------------------------------

#define B_    4
#define N_    2048
#define D_    1024
#define ROWS  (B_ * N_)          // 8192 flat rows
#define SCALE 0.03125f           // 1/sqrt(1024)

typedef __attribute__((ext_vector_type(16))) __bf16 v16bf;
typedef __attribute__((ext_vector_type(8)))  float  v8f;
typedef unsigned int u32x4 __attribute__((ext_vector_type(4)));
typedef int          i32x4 __attribute__((ext_vector_type(4)));
typedef int          i32x8 __attribute__((ext_vector_type(8)));

union BF16x16 { v16bf v; unsigned int u[8]; };

// --- TDM availability probes (compile-safe) --------------------------------
#if defined(__has_builtin)
#  if __has_builtin(__builtin_amdgcn_tensor_load_to_lds)
#    define HAVE_TDM 1
#  endif
#endif
#ifndef HAVE_TDM
#  define HAVE_TDM 0
#endif
#if defined(__has_include)
#  if __has_include(<hip/amd_detail/amd_gfx1250_TDM.h>)
#    define TDM_SIX_ARG 1
#  endif
#endif
#ifndef TDM_SIX_ARG
#  define TDM_SIX_ARG 0
#endif

__device__ __forceinline__ unsigned short f2bf(float f) {
    unsigned int u = __float_as_uint(f);
    return (unsigned short)((u + 0x7FFFu + ((u >> 16) & 1u)) >> 16);  // RNE
}
__device__ __forceinline__ unsigned int pack2(float lo, float hi) {
    return (unsigned int)f2bf(lo) | ((unsigned int)f2bf(hi) << 16);
}

// K-index of the first element of VGPR pair v for a 16-bit 16x32 A operand
// (and symmetric 32x16 B operand), per ISA 7.12.2. Pairs are contiguous in K,
// and each half-group of 4 pairs is 16 contiguous bytes (b128-vectorizable).
__device__ __forceinline__ int kpair(int v, int half) {
    return ((v < 4) ? 2 * v : 2 * v + 8) + half * 8;
}

// A operand (16x32 bf16) from bf16 source (LDS or global), row-major, ld elems.
__device__ __forceinline__ v16bf load_a_bf16(const unsigned short* src, int ld,
                                             int m, int half, int d0) {
    BF16x16 a;
#pragma unroll
    for (int v = 0; v < 8; ++v)
        a.u[v] = *(const unsigned int*)(src + (size_t)m * ld + d0 + kpair(v, half));
    return a.v;
}

// B operand (32x16 bf16), K contiguous in memory: B[k][n] = src[n*ld + d0 + k].
__device__ __forceinline__ v16bf load_b_bf16(const unsigned short* __restrict__ src,
                                             int ld, int n, int half, int d0) {
    BF16x16 b;
#pragma unroll
    for (int v = 0; v < 8; ++v)
        b.u[v] = *(const unsigned int*)(src + (size_t)n * ld + d0 + kpair(v, half));
    return b.v;
}

// --- Tensor Data Mover: 2D bf16 tile (rows x dim0, row stride = stride) -----
// Issued by one wave; EXEC is ignored by TDM. Caller must s_wait_tensorcnt.
__device__ __forceinline__ void tdm_load_tile_bf16(unsigned lds0, const void* gsrc,
                                                   unsigned dim0, unsigned rows,
                                                   unsigned stride) {
#if HAVE_TDM
    unsigned long long ga = (unsigned long long)(size_t)gsrc;
    // D# group0: count=1 | lds_addr | global_addr[56:0] | type=2 ("image")
    u32x4 g0 = { 1u, lds0, (unsigned)ga, (unsigned)(ga >> 32) | (2u << 30) };
    // D# group1: data_size=1 (2B); tensor_dim0=dim0; tensor_dim1=rows;
    //            tile_dim0=dim0; tile_dim1=rows; tensor_dim0_stride=stride.
    i32x8 g1 = { 0x00010000,
                 (int)(dim0 << 16),                    // bits 63:48  dim0 lo
                 (int)((dim0 >> 16) | (rows << 16)),   // dim0 hi | dim1 lo
                 (int)((rows >> 16) | (dim0 << 16)),   // dim1 hi | tile_dim0
                 (int)rows,                            // tile_dim1 | tile_dim2=0
                 (int)stride,                          // dim0_stride[31:0]
                 0, 0 };
    i32x4 gz = { 0, 0, 0, 0 };
#  if TDM_SIX_ARG
    i32x8 gz8 = { 0, 0, 0, 0, 0, 0, 0, 0 };
    __builtin_amdgcn_tensor_load_to_lds(g0, g1, gz, gz, gz8, 0);
#  else
    __builtin_amdgcn_tensor_load_to_lds(g0, g1, gz, gz, 0);
#  endif
#endif
}

__device__ __forceinline__ void tdm_wait0() {
#if HAVE_TDM
#  if defined(__has_builtin) && __has_builtin(__builtin_amdgcn_s_wait_tensorcnt)
    __builtin_amdgcn_s_wait_tensorcnt(0);
#  else
    asm volatile("s_wait_tensorcnt 0x0" ::: "memory");
#  endif
#endif
}

// ---------------------------------------------------------------------------
// Kernel 0: fp32 -> bf16, 4 elements per thread (b128 load, b64 store).
// ---------------------------------------------------------------------------
__global__ __launch_bounds__(256) void f32_to_bf16_kernel(
    const float* __restrict__ in, unsigned short* __restrict__ out, int n4) {
    int i = blockIdx.x * 256 + threadIdx.x;
    if (i < n4) {
        float4 f = ((const float4*)in)[i];
        uint2 o;
        o.x = pack2(f.x, f.y);
        o.y = pack2(f.z, f.w);
        ((uint2*)out)[i] = o;
    }
}

// ---------------------------------------------------------------------------
// Kernel 1: y = x @ W^T (bf16 WMMA), one 16x16 tile per wave.
// A block's 8 waves share one 16-row x tile, staged in LDS by the TDM.
// q,k stored row-major bf16; v stored TRANSPOSED (vt[col][row]) so the
// flash PV B-operand becomes K-contiguous.
// ---------------------------------------------------------------------------
__global__ __launch_bounds__(256) void qkv_gemm_kernel(
    const unsigned short* __restrict__ xb,
    const unsigned short* __restrict__ wq, const unsigned short* __restrict__ wk,
    const unsigned short* __restrict__ wv,
    unsigned short* __restrict__ qb, unsigned short* __restrict__ kb,
    unsigned short* __restrict__ vt) {

    __shared__ unsigned short lX[16 * D_];      // 32 KB: shared x row tile

    const int tid  = threadIdx.x;
    const int lane = tid & 31;
    const int wave = tid >> 5;
    const int half = lane >> 4;
    const int nl   = lane & 15;

    const int tile = blockIdx.x * 8 + wave;     // 8 col tiles of one row tile
    const int r0 = (tile >> 6) * 16;
    const int o0 = (tile & 63) * 16;

    // Stage the 16x1024 bf16 x tile into LDS.
#if HAVE_TDM
    if (wave == 0) {
        tdm_load_tile_bf16((unsigned)(size_t)(void*)&lX[0],
                           xb + (size_t)r0 * D_, D_, 16, D_);
        tdm_wait0();
    }
#else
    {
        const unsigned int* src = (const unsigned int*)(xb + (size_t)r0 * D_);
        unsigned int* dst = (unsigned int*)lX;
        for (int i = tid; i < 16 * D_ / 2; i += 256) dst[i] = src[i];
    }
#endif
    __syncthreads();

    const unsigned short* W = (blockIdx.y == 0) ? wq : (blockIdx.y == 1) ? wk : wv;

    v8f acc = {};
    for (int d0 = 0; d0 < D_; d0 += 32) {
        v16bf a = load_a_bf16(lX, D_, nl, half, d0);
        v16bf b = load_b_bf16(W, D_, o0 + nl, half, d0);
        acc = __builtin_amdgcn_wmma_f32_16x16x32_bf16(false, a, false, b,
                                                      (short)0, acc, false, false);
    }

    if (blockIdx.y < 2) {
        unsigned short* outp = (blockIdx.y == 0) ? qb : kb;
#pragma unroll
        for (int j = 0; j < 8; ++j)
            outp[(size_t)(r0 + j + half * 8) * D_ + o0 + nl] = f2bf(acc[j]);
    } else {
        // Transposed store: vt[(o0+nl)][r0 + half*8 + 0..7] -> one b128 store.
        uint4 pk;
        pk.x = pack2(acc[0], acc[1]);
        pk.y = pack2(acc[2], acc[3]);
        pk.z = pack2(acc[4], acc[5]);
        pk.w = pack2(acc[6], acc[7]);
        *(uint4*)(vt + (size_t)(o0 + nl) * ROWS + r0 + half * 8) = pk;
    }
}

// ---------------------------------------------------------------------------
// Kernel 2: flash attention. One workgroup (8 waves) per 16-query tile.
// Q tile staged into LDS via the Tensor Data Mover; each wave owns a
// 128-column output slice (8 f32 accumulator tiles).
// ---------------------------------------------------------------------------
__global__ __launch_bounds__(256) void flash_attn_kernel(
    const unsigned short* __restrict__ qb,
    const unsigned short* __restrict__ kb,
    const unsigned short* __restrict__ vt,
    float* __restrict__ out) {

    __shared__ unsigned short lQ[16 * D_];   // 32 KB: query tile, bf16
    __shared__ float          lS[16 * 32];   //  2 KB: score tile, fp32
    __shared__ unsigned short lP[16 * 32];   //  1 KB: probability tile, bf16
    __shared__ float lM[16], lL[16], lR[16];

    const int tid  = threadIdx.x;
    const int lane = tid & 31;
    const int wave = tid >> 5;
    const int half = lane >> 4;
    const int nl   = lane & 15;

    const int b     = blockIdx.x >> 7;        // 128 query tiles per batch
    const int q0    = (blockIdx.x & 127) * 16;
    const int rowb  = b * N_;

    // ---- Stage Q tile (16 rows x 1024 cols bf16) into LDS via TDM ----
#if HAVE_TDM
    if (wave == 0) {
        tdm_load_tile_bf16((unsigned)(size_t)(void*)&lQ[0],
                           qb + (size_t)(rowb + q0) * D_, D_, 16, D_);
        tdm_wait0();
    }
#else
    {
        const unsigned int* src = (const unsigned int*)(qb + (size_t)(rowb + q0) * D_);
        unsigned int* dst = (unsigned int*)lQ;
        for (int i = tid; i < 16 * D_ / 2; i += 256) dst[i] = src[i];
    }
#endif
    if (tid < 16) { lM[tid] = -3.0e38f; lL[tid] = 0.0f; }
    __syncthreads();

    const int dslice = wave * 128;            // this wave's D slice

    // Hoist this wave's Q A-operands: loop-invariant across key tiles.
    v16bf aQ[4];
#pragma unroll
    for (int c = 0; c < 4; ++c)
        aQ[c] = load_a_bf16(lQ, D_, nl, half, dslice + c * 32);

    v8f acc[8];
#pragma unroll
    for (int t = 0; t < 8; ++t) acc[t] = (v8f){};

    const int kmax = q0 + 16;                 // causal bound (exclusive)

    for (int j0 = 0; j0 < kmax; j0 += 32) {
        // Prefetch next key tile's K rows for this wave's D slice
        // (global_prefetch_b8: two per lane, 128 B apart, 32 rows covered by 32 lanes).
        if (j0 + 32 < kmax) {
            const unsigned short* pf =
                kb + (size_t)(rowb + j0 + 32 + lane) * D_ + dslice;
            __builtin_prefetch(pf, 0, 3);
            __builtin_prefetch(pf + 64, 0, 3);
        }

        // --- zero the shared score tile ---
        for (int i = tid; i < 512; i += 256) lS[i] = 0.0f;
        __syncthreads();

        // --- partial S = Q[:, slice] @ K[:, slice]^T, two key halves ---
        v8f s0 = {}, s1 = {};
        const unsigned short* k0p = kb + (size_t)(rowb + j0) * D_;
        const unsigned short* k1p = kb + (size_t)(rowb + j0 + 16) * D_;
#pragma unroll
        for (int c = 0; c < 4; ++c) {
            int d0 = dslice + c * 32;
            v16bf b0 = load_b_bf16(k0p, D_, nl, half, d0);
            v16bf b1 = load_b_bf16(k1p, D_, nl, half, d0);
            s0 = __builtin_amdgcn_wmma_f32_16x16x32_bf16(false, aQ[c], false, b0,
                                                         (short)0, s0, false, false);
            s1 = __builtin_amdgcn_wmma_f32_16x16x32_bf16(false, aQ[c], false, b1,
                                                         (short)0, s1, false, false);
        }
#pragma unroll
        for (int j = 0; j < 8; ++j) {
            int row = j + half * 8;
            atomicAdd(&lS[row * 32 + nl],      s0[j]);   // ds_add_f32
            atomicAdd(&lS[row * 32 + 16 + nl], s1[j]);
        }
        __syncthreads();

        // --- online softmax over this 16x32 score tile (threads 0..15) ---
        if (tid < 16) {
            const int row  = tid;
            const int qrow = q0 + row;
            int vcols = qrow - j0 + 1;
            if (vcols > 32) vcols = 32;
            if (vcols < 0)  vcols = 0;
            float m_old = lM[row];
            float mx = m_old;
            for (int c = 0; c < vcols; ++c)
                mx = fmaxf(mx, lS[row * 32 + c] * SCALE);
            float resc = __expf(m_old - mx);
            float sum = 0.0f;
            for (int c = 0; c < 32; ++c) {
                float p = (c < vcols) ? __expf(lS[row * 32 + c] * SCALE - mx) : 0.0f;
                sum += p;
                lP[row * 32 + c] = f2bf(p);
            }
            lL[row] = lL[row] * resc + sum;
            lM[row] = mx;
            lR[row] = resc;
        }
        __syncthreads();

        // --- rescale accumulators, then O += P @ V for this wave's columns ---
        float rsc[8];
#pragma unroll
        for (int j = 0; j < 8; ++j) rsc[j] = lR[j + half * 8];

        v16bf ap = load_a_bf16(lP, 32, nl, half, 0);   // P is 16x32 bf16
#pragma unroll
        for (int t = 0; t < 8; ++t) {
            int col = dslice + t * 16 + nl;
            const unsigned short* vcol = vt + (size_t)col * ROWS + rowb + j0;
            // Prefetch this column's V rows for the next key tile.
            if (j0 + 32 < kmax) __builtin_prefetch(vcol + 32, 0, 3);
            // B[k][n] = V[j0+k][col] = vt[col][rowb+j0+k]: K-contiguous pairs.
            BF16x16 bv;
#pragma unroll
            for (int v = 0; v < 8; ++v)
                bv.u[v] = *(const unsigned int*)(vcol + kpair(v, half));
#pragma unroll
            for (int j = 0; j < 8; ++j) acc[t][j] *= rsc[j];
            acc[t] = __builtin_amdgcn_wmma_f32_16x16x32_bf16(false, ap, false, bv.v,
                                                             (short)0, acc[t], false, false);
        }
        __syncthreads();   // protect lS/lP before next iteration
    }

    // --- epilogue: divide by row sum, write fp32 output ---
    float linv[8];
#pragma unroll
    for (int j = 0; j < 8; ++j) linv[j] = 1.0f / lL[j + half * 8];
#pragma unroll
    for (int t = 0; t < 8; ++t) {
        int col0 = dslice + t * 16;
#pragma unroll
        for (int j = 0; j < 8; ++j) {
            int row = rowb + q0 + j + half * 8;
            out[(size_t)row * D_ + col0 + nl] = acc[t][j] * linv[j];
        }
    }
}

// ---------------------------------------------------------------------------
extern "C" void kernel_launch(void* const* d_in, const int* in_sizes, int n_in,
                              void* d_out, int out_size, void* d_ws, size_t ws_size,
                              hipStream_t stream) {
    const float* x  = (const float*)d_in[0];
    const float* Wq = (const float*)d_in[1];
    const float* Wk = (const float*)d_in[2];
    const float* Wv = (const float*)d_in[3];

    const size_t xe = (size_t)ROWS * D_;     // 8,388,608
    const size_t we = (size_t)D_ * D_;       // 1,048,576
    unsigned short* xb  = (unsigned short*)d_ws;
    unsigned short* wqb = xb  + xe;
    unsigned short* wkb = wqb + we;
    unsigned short* wvb = wkb + we;
    unsigned short* qb  = wvb + we;
    unsigned short* kb  = qb  + xe;
    unsigned short* vt  = kb  + xe;          // total ~73.4 MB of workspace

    // fp32 -> bf16 conversions (4 elems/thread).
    f32_to_bf16_kernel<<<dim3((unsigned)(xe / 4 / 256)), 256, 0, stream>>>(x,  xb,  (int)(xe / 4));
    f32_to_bf16_kernel<<<dim3((unsigned)(we / 4 / 256)), 256, 0, stream>>>(Wq, wqb, (int)(we / 4));
    f32_to_bf16_kernel<<<dim3((unsigned)(we / 4 / 256)), 256, 0, stream>>>(Wk, wkb, (int)(we / 4));
    f32_to_bf16_kernel<<<dim3((unsigned)(we / 4 / 256)), 256, 0, stream>>>(Wv, wvb, (int)(we / 4));

    // QKV projections: 32768 tiles per matrix, 8 tiles (waves) per block.
    qkv_gemm_kernel<<<dim3(32768 / 8, 3), 256, 0, stream>>>(
        xb, wqb, wkb, wvb, qb, kb, vt);

    // Flash attention: one block per (batch, 16-query tile).
    flash_attn_kernel<<<dim3(B_ * (N_ / 16)), 256, 0, stream>>>(
        qb, kb, vt, (float*)d_out);
}
